// GradGNN_79577154060662
// MI455X (gfx1250) — matrix operand order repaired and checked by reading, not compiled, not run on hardware
//
#include <hip/hip_runtime.h>

#define NHEADS 4
#define CDIM 128
#define HCDIM 512
#define INDIM 256
#define OUTDIM 40
#define NEG_SLOPE 0.2f

typedef __attribute__((ext_vector_type(16))) _Float16 v16h;
typedef __attribute__((ext_vector_type(8)))  _Float16 v8h;
typedef __attribute__((ext_vector_type(8)))  float    v8f;
typedef __attribute__((ext_vector_type(4)))  float    v4f;

// ---------- helpers ----------
__device__ __forceinline__ unsigned enc_f(float f) {
    unsigned u = __float_as_uint(f);
    return (u & 0x80000000u) ? ~u : (u | 0x80000000u);
}
__device__ __forceinline__ float dec_f(unsigned u) {
    unsigned b = (u & 0x80000000u) ? (u & 0x7fffffffu) : ~u;
    return __uint_as_float(b);
}
__device__ __forceinline__ void get_edge(const int* __restrict__ ei, int E, int eidx,
                                         int& s, int& d) {
    if (eidx < E) { s = ei[eidx]; d = ei[E + eidx]; }
    else          { s = eidx - E; d = eidx - E; }
}

// ---------- utility kernels ----------
__global__ void k_zero_u32(unsigned* __restrict__ p, int n) {
    int i = blockIdx.x * blockDim.x + threadIdx.x;
    if (i < n) p[i] = 0u;
}

__global__ void k_f32_to_f16(const float* __restrict__ in, _Float16* __restrict__ out, int n) {
    int i = blockIdx.x * blockDim.x + threadIdx.x;
    if (i < n) out[i] = (_Float16)in[i];
}

// W is (K x Nc) row-major fp32; out is (Nc x K) row-major f16 (transposed copy)
__global__ void k_transpose_h(const float* __restrict__ W, _Float16* __restrict__ out,
                              int K, int Nc) {
    int i = blockIdx.x * blockDim.x + threadIdx.x;
    if (i >= K * Nc) return;
    int k = i / Nc, n = i % Nc;
    out[(size_t)n * K + k] = (_Float16)W[i];
}

// ---------- WMMA GEMM ----------
// C[M x NcReal] = A(M x K f16) * B(K x Nc f16, stored transposed: NcPad x K) + bias.
// One wave computes a 16 x (NB*16) output strip: A fragment loaded once per k-step
// and reused across NB B fragments / NB accumulators (fragment traffic ~ (1+NB)/NB
// per WMMA instead of 2). K multiple of 32; M multiple of 16.
template <int NB>
__global__ void k_wmma_gemm(const _Float16* __restrict__ A,
                            const _Float16* __restrict__ BT,
                            const float* __restrict__ bias,
                            float* __restrict__ C,
                            int K, int nblk, int ntiles, int NcReal, int ldc, int act) {
    int wv = blockIdx.x * (blockDim.x >> 5) + (threadIdx.x >> 5);
    if (wv >= ntiles) return;
    int tm = wv / nblk, tnb = wv % nblk;
    int lane = threadIdx.x & 31;
    int l15 = lane & 15;
    bool hi = lane >= 16;
    const _Float16* arow  = A  + (size_t)(tm * 16 + l15) * K;
    const _Float16* brow0 = BT + (size_t)(tnb * NB * 16 + l15) * K;
    int aoff = hi ? 16 : 0;   // A: low lanes K{0..15}, high lanes K{16..31}
    int boff = hi ? 16 : 0;   // B: low lanes K{0..15}, high lanes K{16..31}
    v8f acc[NB];
#pragma unroll
    for (int j = 0; j < NB; ++j) acc[j] = (v8f){};
    for (int k0 = 0; k0 < K; k0 += 32) {
        v16h av = *(const v16h*)(arow + k0 + aoff);
#pragma unroll
        for (int j = 0; j < NB; ++j) {
            v16h bv = *(const v16h*)(brow0 + (size_t)j * 16 * K + k0 + boff);
            acc[j] = __builtin_amdgcn_wmma_f32_16x16x32_f16(false, av, false, bv,
                                                            (short)0, acc[j], false, false);
        }
    }
    int mbase = tm * 16 + (hi ? 8 : 0);
#pragma unroll
    for (int j = 0; j < NB; ++j) {
        int n = (tnb * NB + j) * 16 + l15;
        if (n >= NcReal) continue;
        float b = bias ? bias[n] : 0.f;
#pragma unroll
        for (int v = 0; v < 8; ++v) {
            float val = acc[j][v] + b;
            if (act) val = fmaxf(val, 0.f);
            C[(size_t)(mbase + v) * ldc + n] = val;
        }
    }
}

// ---------- edge phase ----------
// e[eidx,h] = sum_c att[h,c] * leaky_relu(xl[src,h,c] + xr[dst,h,c])
__global__ void k_edge_e(const float* __restrict__ xl, const float* __restrict__ xr,
                         const float* __restrict__ att, const int* __restrict__ ei,
                         int E, float* __restrict__ e, int Etot) {
    int gid = blockIdx.x * blockDim.x + threadIdx.x;
    if (gid >= Etot * NHEADS) return;
    int h = gid & 3, eidx = gid >> 2;
    int s, d; get_edge(ei, E, eidx, s, d);
    const float* pl = xl + (size_t)s * HCDIM + h * CDIM;
    const float* pr = xr + (size_t)d * HCDIM + h * CDIM;
    const float* pa = att + h * CDIM;
    float acc = 0.f;
    for (int c = 0; c < CDIM; c += 4) {
        v4f a = *(const v4f*)(pl + c);
        v4f b = *(const v4f*)(pr + c);
        v4f w = *(const v4f*)(pa + c);
#pragma unroll
        for (int j = 0; j < 4; ++j) {
            float f = a[j] + b[j];
            f = f > 0.f ? f : NEG_SLOPE * f;
            acc += w[j] * f;
        }
    }
    e[gid] = acc;
}

__global__ void k_segmax(const float* __restrict__ e, const int* __restrict__ ei,
                         int E, unsigned* __restrict__ m, int Etot) {
    int gid = blockIdx.x * blockDim.x + threadIdx.x;
    if (gid >= Etot * NHEADS) return;
    int h = gid & 3, eidx = gid >> 2;
    int s, d; get_edge(ei, E, eidx, s, d);
    atomicMax(&m[(size_t)d * NHEADS + h], enc_f(e[gid]));
}

__global__ void k_expsum(float* __restrict__ e, const int* __restrict__ ei, int E,
                         const unsigned* __restrict__ m, float* __restrict__ s, int Etot) {
    int gid = blockIdx.x * blockDim.x + threadIdx.x;
    if (gid >= Etot * NHEADS) return;
    int h = gid & 3, eidx = gid >> 2;
    int sr, d; get_edge(ei, E, eidx, sr, d);
    float mv = dec_f(m[(size_t)d * NHEADS + h]);
    float ex = expf(e[gid] - mv);
    e[gid] = ex;
    atomicAdd(&s[(size_t)d * NHEADS + h], ex);
}

// agg[dst, h, c..c+3] += xl[src, h, c..c+3] * (ex / (s[dst,h] + 1e-16))
__global__ void k_aggregate(const float* __restrict__ xl, const float* __restrict__ e,
                            const float* __restrict__ s, const int* __restrict__ ei,
                            int E, float* __restrict__ agg, long long total) {
    long long gid = blockIdx.x * (long long)blockDim.x + threadIdx.x;
    if (gid >= total) return;
    int c4 = (int)(gid & 31);
    int h = (int)((gid >> 5) & 3);
    int eidx = (int)(gid >> 7);
    int sr, d; get_edge(ei, E, eidx, sr, d);
    float alpha = e[(size_t)eidx * NHEADS + h] / (s[(size_t)d * NHEADS + h] + 1e-16f);
    const float* px = xl + (size_t)sr * HCDIM + h * CDIM + c4 * 4;
    float* pa = agg + (size_t)d * HCDIM + h * CDIM + c4 * 4;
    v4f xv = *(const v4f*)px;
    atomicAdd(pa + 0, xv[0] * alpha);
    atomicAdd(pa + 1, xv[1] * alpha);
    atomicAdd(pa + 2, xv[2] * alpha);
    atomicAdd(pa + 3, xv[3] * alpha);
}

// h1 = relu(agg + bo1), written directly as f16 for the next WMMA stage
__global__ void k_finalize1(const float* __restrict__ agg, const float* __restrict__ bo,
                            _Float16* __restrict__ h1h, int total) {
    int i = blockIdx.x * blockDim.x + threadIdx.x;
    if (i >= total) return;
    float v = fmaxf(agg[i] + bo[i & (HCDIM - 1)], 0.f);
    h1h[i] = (_Float16)v;
}

// h2 = relu(mean_heads(agg) + bo2), as f16
__global__ void k_finalize2(const float* __restrict__ agg, const float* __restrict__ bo,
                            _Float16* __restrict__ h2h, int total) {
    int i = blockIdx.x * blockDim.x + threadIdx.x;
    if (i >= total) return;
    int n = i >> 7, c = i & (CDIM - 1);
    const float* p = agg + (size_t)n * HCDIM + c;
    float v = (p[0] + p[CDIM] + p[2 * CDIM] + p[3 * CDIM]) * 0.25f + bo[c];
    h2h[i] = (_Float16)fmaxf(v, 0.f);
}

// z = z2 / max(||z2||, 1e-12); one wave per row (128 cols = 32 lanes x float4)
__global__ void k_normalize(const float* __restrict__ z, float* __restrict__ out, int N) {
    int wv = blockIdx.x * (blockDim.x >> 5) + (threadIdx.x >> 5);
    int lane = threadIdx.x & 31;
    if (wv >= N) return;
    const float* row = z + (size_t)wv * CDIM;
    v4f v = *(const v4f*)(row + lane * 4);
    float ss = v[0] * v[0] + v[1] * v[1] + v[2] * v[2] + v[3] * v[3];
#pragma unroll
    for (int o = 16; o; o >>= 1) ss += __shfl_xor(ss, o, 32);
    float inv = 1.f / fmaxf(sqrtf(ss), 1e-12f);
    v4f r = { v[0] * inv, v[1] * inv, v[2] * inv, v[3] * inv };
    *(v4f*)(out + (size_t)wv * CDIM + lane * 4) = r;
}

// ---------- launch ----------
extern "C" void kernel_launch(void* const* d_in, const int* in_sizes, int n_in,
                              void* d_out, int out_size, void* d_ws, size_t ws_size,
                              hipStream_t stream) {
    const float* x    = (const float*)d_in[0];
    const int*   ei   = (const int*)d_in[1];
    const float* Wl1  = (const float*)d_in[2];  const float* bl1 = (const float*)d_in[3];
    const float* Wr1  = (const float*)d_in[4];  const float* br1 = (const float*)d_in[5];
    const float* att1 = (const float*)d_in[6];  const float* bo1 = (const float*)d_in[7];
    const float* Wl2  = (const float*)d_in[8];  const float* bl2 = (const float*)d_in[9];
    const float* Wr2  = (const float*)d_in[10]; const float* br2 = (const float*)d_in[11];
    const float* att2 = (const float*)d_in[12]; const float* bo2 = (const float*)d_in[13];
    const float* Wp1  = (const float*)d_in[14]; const float* bp1 = (const float*)d_in[15];
    const float* Wp2  = (const float*)d_in[16]; const float* bp2 = (const float*)d_in[17];
    const float* Wc   = (const float*)d_in[18]; const float* bc  = (const float*)d_in[19];

    const int N = in_sizes[0] / INDIM;
    const int E = in_sizes[1] / 2;
    const int Etot = E + N;

    char* ws = (char*)d_ws;
    size_t off = 0;
    auto alloc = [&](size_t bytes) -> void* {
        void* p = ws + off;
        off += (bytes + 255) & ~(size_t)255;
        return p;
    };
    _Float16* xh   = (_Float16*)alloc((size_t)N * INDIM * 2);
    _Float16* h1h  = (_Float16*)alloc((size_t)N * HCDIM * 2);
    _Float16* h2h  = (_Float16*)alloc((size_t)N * CDIM * 2);
    _Float16* z1h  = (_Float16*)alloc((size_t)N * CDIM * 2);
    _Float16* wlt1 = (_Float16*)alloc((size_t)HCDIM * INDIM * 2);
    _Float16* wrt1 = (_Float16*)alloc((size_t)HCDIM * INDIM * 2);
    _Float16* wlt2 = (_Float16*)alloc((size_t)HCDIM * HCDIM * 2);
    _Float16* wrt2 = (_Float16*)alloc((size_t)HCDIM * HCDIM * 2);
    _Float16* wp1t = (_Float16*)alloc((size_t)CDIM * CDIM * 2);
    _Float16* wp2t = (_Float16*)alloc((size_t)CDIM * CDIM * 2);
    _Float16* wct  = (_Float16*)alloc((size_t)48 * CDIM * 2);    // 40 cols padded to 48
    float*    xl   = (float*)alloc((size_t)N * HCDIM * 4);
    float*    xr   = (float*)alloc((size_t)N * HCDIM * 4);
    float*    agg  = (float*)alloc((size_t)N * HCDIM * 4);
    float*    ebuf = (float*)alloc((size_t)Etot * NHEADS * 4);
    unsigned* mbuf = (unsigned*)alloc((size_t)N * NHEADS * 4);
    float*    sbuf = (float*)alloc((size_t)N * NHEADS * 4);
    float*    z1f  = xl;   // GEMM scratch reuse after edge phases are done
    float*    z2f  = xr;

    float* out_cls = (float*)d_out;                       // N x 40
    float* out_z   = (float*)d_out + (size_t)N * OUTDIM;  // N x 128

    const int T = 256;
    auto B = [](long long n) { return (unsigned)((n + 255) / 256); };
    auto GB = [](int ntiles) { return (unsigned)((ntiles + 7) / 8); }; // 8 waves/block

    const int mt = N / 16;  // 3125, exact

    // --- convert activations / weights ---
    k_f32_to_f16<<<B((long long)N * INDIM), T, 0, stream>>>(x, xh, N * INDIM);
    k_transpose_h<<<B((long long)INDIM * HCDIM), T, 0, stream>>>(Wl1, wlt1, INDIM, HCDIM);
    k_transpose_h<<<B((long long)INDIM * HCDIM), T, 0, stream>>>(Wr1, wrt1, INDIM, HCDIM);
    k_transpose_h<<<B((long long)HCDIM * HCDIM), T, 0, stream>>>(Wl2, wlt2, HCDIM, HCDIM);
    k_transpose_h<<<B((long long)HCDIM * HCDIM), T, 0, stream>>>(Wr2, wrt2, HCDIM, HCDIM);
    k_transpose_h<<<B((long long)CDIM * CDIM), T, 0, stream>>>(Wp1, wp1t, CDIM, CDIM);
    k_transpose_h<<<B((long long)CDIM * CDIM), T, 0, stream>>>(Wp2, wp2t, CDIM, CDIM);
    k_zero_u32<<<B(48 * CDIM / 2), T, 0, stream>>>((unsigned*)wct, 48 * CDIM / 2);
    k_transpose_h<<<B((long long)CDIM * OUTDIM), T, 0, stream>>>(Wc, wct, CDIM, OUTDIM);

    // ======== layer 1: xl = x@Wl1+bl1, xr = x@Wr1+br1 ========
    {
        int nblk = (HCDIM / 16) / 4, ntiles = mt * nblk;   // NB=4 -> 8 n-blocks
        k_wmma_gemm<4><<<GB(ntiles), T, 0, stream>>>(xh, wlt1, bl1, xl, INDIM, nblk, ntiles, HCDIM, HCDIM, 0);
        k_wmma_gemm<4><<<GB(ntiles), T, 0, stream>>>(xh, wrt1, br1, xr, INDIM, nblk, ntiles, HCDIM, HCDIM, 0);
    }
    k_edge_e<<<B((long long)Etot * NHEADS), T, 0, stream>>>(xl, xr, att1, ei, E, ebuf, Etot);
    k_zero_u32<<<B((long long)N * NHEADS), T, 0, stream>>>(mbuf, N * NHEADS);
    k_segmax<<<B((long long)Etot * NHEADS), T, 0, stream>>>(ebuf, ei, E, mbuf, Etot);
    k_zero_u32<<<B((long long)N * NHEADS), T, 0, stream>>>((unsigned*)sbuf, N * NHEADS);
    k_expsum<<<B((long long)Etot * NHEADS), T, 0, stream>>>(ebuf, ei, E, mbuf, sbuf, Etot);
    k_zero_u32<<<B((long long)N * HCDIM), T, 0, stream>>>((unsigned*)agg, N * HCDIM);
    k_aggregate<<<B((long long)Etot * 128), T, 0, stream>>>(xl, ebuf, sbuf, ei, E, agg, (long long)Etot * 128);
    k_finalize1<<<B((long long)N * HCDIM), T, 0, stream>>>(agg, bo1, h1h, N * HCDIM);

    // ======== layer 2 ========
    {
        int nblk = (HCDIM / 16) / 4, ntiles = mt * nblk;
        k_wmma_gemm<4><<<GB(ntiles), T, 0, stream>>>(h1h, wlt2, bl2, xl, HCDIM, nblk, ntiles, HCDIM, HCDIM, 0);
        k_wmma_gemm<4><<<GB(ntiles), T, 0, stream>>>(h1h, wrt2, br2, xr, HCDIM, nblk, ntiles, HCDIM, HCDIM, 0);
    }
    k_edge_e<<<B((long long)Etot * NHEADS), T, 0, stream>>>(xl, xr, att2, ei, E, ebuf, Etot);
    k_zero_u32<<<B((long long)N * NHEADS), T, 0, stream>>>(mbuf, N * NHEADS);
    k_segmax<<<B((long long)Etot * NHEADS), T, 0, stream>>>(ebuf, ei, E, mbuf, Etot);
    k_zero_u32<<<B((long long)N * NHEADS), T, 0, stream>>>((unsigned*)sbuf, N * NHEADS);
    k_expsum<<<B((long long)Etot * NHEADS), T, 0, stream>>>(ebuf, ei, E, mbuf, sbuf, Etot);
    k_zero_u32<<<B((long long)N * HCDIM), T, 0, stream>>>((unsigned*)agg, N * HCDIM);
    k_aggregate<<<B((long long)Etot * 128), T, 0, stream>>>(xl, ebuf, sbuf, ei, E, agg, (long long)Etot * 128);
    k_finalize2<<<B((long long)N * CDIM), T, 0, stream>>>(agg, bo2, h2h, N * CDIM);

    // ======== MLP head (z) ========
    {
        int nblk = (CDIM / 16) / 4, ntiles = mt * nblk;    // 2 n-blocks
        k_wmma_gemm<4><<<GB(ntiles), T, 0, stream>>>(h2h, wp1t, bp1, z1f, CDIM, nblk, ntiles, CDIM, CDIM, 1);
        k_f32_to_f16<<<B((long long)N * CDIM), T, 0, stream>>>(z1f, z1h, N * CDIM);
        k_wmma_gemm<4><<<GB(ntiles), T, 0, stream>>>(z1h, wp2t, bp2, z2f, CDIM, nblk, ntiles, CDIM, CDIM, 0);
        k_normalize<<<B((long long)N * 32), T, 0, stream>>>(z2f, out_z, N);
    }

    // ======== classifier head (out): 48-padded, NB=3 covers the whole row ========
    {
        int nblk = 1, ntiles = mt;
        k_wmma_gemm<3><<<GB(ntiles), T, 0, stream>>>(h2h, wct, bc, out_cls, CDIM, nblk, ntiles, OUTDIM, OUTDIM, 0);
    }
}
